// Network_48017734369861
// MI455X (gfx1250) — matrix operand order repaired and checked by reading, not compiled, not run on hardware
//
#include <hip/hip_runtime.h>

// ---------------------------------------------------------------------------
// Adaptive log-softmax with MLP front-end, fused for gfx1250 (MI455X).
// All GEMMs run on v_wmma_f32_16x16x32_bf16. The big logit matrices
// (4096x2002 / 4096x8000 / 4096x40000) are never materialized: each 16-row
// block streams 16x16 logit tiles out of the WMMA accumulators and maintains
// an online (max, sumexp) -> per-row logsumexp.
//   - LSE kernel is 1x4 N-register-blocked: A-fragment loaded once per K-step,
//     reused across 4 B tiles / 4 independent accumulators.
//   - GEMM kernel is 4x1 M-register-blocked: B-fragment (weights) reused
//     across 4 A tiles.
// Gathered target logits come from cheap per-row dot products vs f32 weights.
// ---------------------------------------------------------------------------

typedef __attribute__((ext_vector_type(16))) __bf16 v16bf;
typedef __attribute__((ext_vector_type(8)))  float  v8f;

union Frag {
  v16bf v;
  uint4 u[2];
};

__device__ __forceinline__ unsigned short f32_to_bf16(float f) {
  unsigned int u = __float_as_uint(f);
  u += 0x7FFFu + ((u >> 16) & 1u);          // round-to-nearest-even
  return (unsigned short)(u >> 16);
}
__device__ __forceinline__ float bf16_to_f32(unsigned short h) {
  return __uint_as_float(((unsigned int)h) << 16);
}

// ---------------------------------------------------------------------------
// f32 -> bf16 conversion (zero-pads tail up to ntot)
// ---------------------------------------------------------------------------
__global__ void cvt_kernel(const float* __restrict__ src,
                           unsigned short* __restrict__ dst, int n, int ntot) {
  int i = blockIdx.x * 256 + threadIdx.x;
  if (i < ntot) dst[i] = (i < n) ? f32_to_bf16(src[i]) : (unsigned short)0;
}

// ---------------------------------------------------------------------------
// D[M,N] = act( A[M,K] @ W[N,K]^T + bias ), output bf16.
// One wave computes a 64x16 strip (4 M-tiles); B-fragment reused 4x.
// Fragment layouts per CDNA5 ISA:
//   A (16-bit 16x32): lane<16 holds {K0-7,K16-23}, lane>=16 {K8-15,K24-31}
//   B (16-bit 32x16): same K order as A: lane<16 {K0-7,K16-23}, lane>=16 {K8-15,K24-31}
//   C/D: element (vgpr v, lane L) = row v + 8*(L>>4), col L&15
// ---------------------------------------------------------------------------
__global__ void gemm_bf16_kernel(const unsigned short* __restrict__ A,
                                 const unsigned short* __restrict__ W,
                                 const float* __restrict__ bias,
                                 unsigned short* __restrict__ D,
                                 int M, int N, int K, int relu) {
  const int wave = threadIdx.x >> 5;
  const int lane = threadIdx.x & 31;
  const int tiles_n = N >> 4;
  const int total = (M >> 6) * tiles_n;        // 64-row strips
  const int wid = blockIdx.x * 8 + wave;
  if (wid >= total) return;                    // wave-uniform exit, EXEC stays full
  const int bm = wid / tiles_n, tn = wid % tiles_n;
  const int lsub = lane & 15, half = lane >> 4;

  const unsigned short* arow = A + (bm * 64 + lsub) * K + half * 8;
  const unsigned short* wrow = W + (tn * 16 + lsub) * K + half * 8;

  v8f acc[4] = {};
  for (int k0 = 0; k0 < K; k0 += 32) {
    Frag b;
    b.u[0] = *(const uint4*)(wrow + k0);
    b.u[1] = *(const uint4*)(wrow + k0 + 16);
#pragma unroll
    for (int i = 0; i < 4; ++i) {
      Frag a;
      const unsigned short* ai = arow + i * 16 * K;
      a.u[0] = *(const uint4*)(ai + k0);
      a.u[1] = *(const uint4*)(ai + k0 + 16);
      acc[i] = __builtin_amdgcn_wmma_f32_16x16x32_bf16(
          false, a.v, false, b.v, (short)0, acc[i], false, false);
    }
  }

  const int col = tn * 16 + lsub;
  const float bv = bias ? bias[col] : 0.f;
#pragma unroll
  for (int i = 0; i < 4; ++i) {
    const int rbase = bm * 64 + i * 16 + half * 8;
#pragma unroll
    for (int v = 0; v < 8; ++v) {
      float val = acc[i][v] + bv;
      if (relu) val = fmaxf(val, 0.f);
      D[(rbase + v) * N + col] = f32_to_bf16(val);
    }
  }
}

// ---------------------------------------------------------------------------
// Fused logits + online logsumexp. One block = 16 rows; 8 waves stride over
// groups of 4 N-tiles (A-fragment reused 4x per K-step); LDS merge at end.
// Weight matrices are zero-padded so NtilesPad % 32 == 0; columns >= Nreal
// are masked to -inf before the max/exp.
// ---------------------------------------------------------------------------
__global__ void lse_kernel(const unsigned short* __restrict__ A,
                           const unsigned short* __restrict__ W,
                           const float* __restrict__ bias,
                           float* __restrict__ lse,
                           int K, int Nreal, int NtilesPad) {
  __shared__ float sm[8 * 16];
  __shared__ float ss[8 * 16];
  const int wave = threadIdx.x >> 5;
  const int lane = threadIdx.x & 31;
  const int lsub = lane & 15, half = lane >> 4;
  const int row0 = blockIdx.x * 16;
  const unsigned short* arow = A + (row0 + lsub) * K + half * 8;

  float rm[8], rs[8];
#pragma unroll
  for (int v = 0; v < 8; ++v) { rm[v] = -INFINITY; rs[v] = 0.f; }

  for (int tbase = wave * 4; tbase < NtilesPad; tbase += 32) {
    const unsigned short* wrow = W + (tbase * 16 + lsub) * K + half * 8;
    v8f acc[4] = {};
    for (int k0 = 0; k0 < K; k0 += 32) {
      Frag a;
      a.u[0] = *(const uint4*)(arow + k0);
      a.u[1] = *(const uint4*)(arow + k0 + 16);
#pragma unroll
      for (int j = 0; j < 4; ++j) {
        Frag b;
        const unsigned short* wj = wrow + j * 16 * K;
        b.u[0] = *(const uint4*)(wj + k0);
        b.u[1] = *(const uint4*)(wj + k0 + 16);
        acc[j] = __builtin_amdgcn_wmma_f32_16x16x32_bf16(
            false, a.v, false, b.v, (short)0, acc[j], false, false);
      }
    }
#pragma unroll
    for (int j = 0; j < 4; ++j) {
      const int n = (tbase + j) * 16 + lsub;
      const bool valid = n < Nreal;
      const float bv = (bias && valid) ? bias[n] : 0.f;
#pragma unroll
      for (int v = 0; v < 8; ++v) {
        // row = v + 8*half lives across the 16 lanes of this half
        float logit = valid ? acc[j][v] + bv : -INFINITY;
        float tmax = logit;
#pragma unroll
        for (int m = 8; m >= 1; m >>= 1) tmax = fmaxf(tmax, __shfl_xor(tmax, m, 32));
        float e = __expf(logit - tmax);       // exp(-inf - finite) = 0
#pragma unroll
        for (int m = 8; m >= 1; m >>= 1) e += __shfl_xor(e, m, 32);
        const float nm = fmaxf(rm[v], tmax);
        rs[v] = rs[v] * __expf(rm[v] - nm) + e * __expf(tmax - nm);
        rm[v] = nm;
      }
    }
  }

  if (lsub == 0) {
#pragma unroll
    for (int v = 0; v < 8; ++v) {
      sm[wave * 16 + half * 8 + v] = rm[v];
      ss[wave * 16 + half * 8 + v] = rs[v];
    }
  }
  __syncthreads();
  if (threadIdx.x < 16) {
    const int r = threadIdx.x;
    float m = -INFINITY;
    for (int w = 0; w < 8; ++w) m = fmaxf(m, sm[w * 16 + r]);
    float s = 0.f;
    for (int w = 0; w < 8; ++w) s += ss[w * 16 + r] * __expf(sm[w * 16 + r] - m);
    lse[row0 + r] = m + __logf(s);
  }
}

// ---------------------------------------------------------------------------
// Per-row gathered logits (dot products vs original f32 weights) + final
// select; also block-partial sums for the mean (deterministic, no atomics).
// ---------------------------------------------------------------------------
__global__ void final_kernel(const unsigned short* __restrict__ scoreb,
                             const unsigned short* __restrict__ u0b,
                             const unsigned short* __restrict__ u1b,
                             const float* __restrict__ head_W,
                             const float* __restrict__ head_b,
                             const float* __restrict__ t0b_w,
                             const float* __restrict__ t1b_w,
                             const int* __restrict__ tgt,
                             const float* __restrict__ lse_h,
                             const float* __restrict__ lse0,
                             const float* __restrict__ lse1,
                             float* __restrict__ out,
                             float* __restrict__ partial) {
  __shared__ float psum[8];
  const int wave = threadIdx.x >> 5;
  const int lane = threadIdx.x & 31;
  const int row = blockIdx.x * 8 + wave;
  const int t = tgt[row];
  const int th = min(max(t, 0), 1999);
  const int i0 = min(max(t - 2000, 0), 7999);
  const int i1 = min(max(t - 10000, 0), 39999);

  float dh = 0.f, d20 = 0.f, d21 = 0.f;
  const unsigned short* s = scoreb + row * 512;
  for (int k = lane; k < 512; k += 32) {
    const float sv = bf16_to_f32(s[k]);
    dh  += sv * head_W[th * 512 + k];
    d20 += sv * head_W[2000 * 512 + k];
    d21 += sv * head_W[2001 * 512 + k];
  }
  float g0 = 0.f;
  const unsigned short* u0 = u0b + row * 128;
  for (int k = lane; k < 128; k += 32)
    g0 += bf16_to_f32(u0[k]) * t0b_w[i0 * 128 + k];
  float g1 = bf16_to_f32(u1b[row * 32 + lane]) * t1b_w[i1 * 32 + lane];

#pragma unroll
  for (int m = 16; m >= 1; m >>= 1) {
    dh  += __shfl_xor(dh, m, 32);
    d20 += __shfl_xor(d20, m, 32);
    d21 += __shfl_xor(d21, m, 32);
    g0  += __shfl_xor(g0, m, 32);
    g1  += __shfl_xor(g1, m, 32);
  }

  if (lane == 0) {
    const float lh = lse_h[row];
    const float lp_short = dh + head_b[th] - lh;
    const float lp0 = (d20 + head_b[2000] - lh) + (g0 - lse0[row]);
    const float lp1 = (d21 + head_b[2001] - lh) + (g1 - lse1[row]);
    const float o = (t < 2000) ? lp_short : ((t < 10000) ? lp0 : lp1);
    out[row] = o;
    psum[wave] = o;
  }
  __syncthreads();
  if (threadIdx.x == 0) {
    float sum = 0.f;
    for (int w = 0; w < 8; ++w) sum += psum[w];
    partial[blockIdx.x] = sum;
  }
}

__global__ void loss_kernel(const float* __restrict__ partial, int np,
                            float* __restrict__ out_loss) {
  __shared__ float sh[256];
  float s = 0.f;
  for (int i = threadIdx.x; i < np; i += 256) s += partial[i];
  sh[threadIdx.x] = s;
  __syncthreads();
  for (int off = 128; off >= 1; off >>= 1) {
    if ((int)threadIdx.x < off) sh[threadIdx.x] += sh[threadIdx.x + off];
    __syncthreads();
  }
  if (threadIdx.x == 0) out_loss[0] = -sh[0] / 4096.f;
}

// ---------------------------------------------------------------------------
extern "C" void kernel_launch(void* const* d_in, const int* in_sizes, int n_in,
                              void* d_out, int out_size, void* d_ws, size_t ws_size,
                              hipStream_t stream) {
  (void)in_sizes; (void)n_in; (void)out_size; (void)ws_size;
  const float* x      = (const float*)d_in[0];
  const int*   t      = (const int*)  d_in[1];
  const float* W1     = (const float*)d_in[2];
  const float* b1     = (const float*)d_in[3];
  const float* W2     = (const float*)d_in[4];
  const float* b2     = (const float*)d_in[5];
  const float* W3     = (const float*)d_in[6];
  const float* b3     = (const float*)d_in[7];
  const float* head_W = (const float*)d_in[8];
  const float* head_b = (const float*)d_in[9];
  const float* t0a    = (const float*)d_in[10];
  const float* t0bw   = (const float*)d_in[11];
  const float* t1a    = (const float*)d_in[12];
  const float* t1bw   = (const float*)d_in[13];
  float* out = (float*)d_out;                 // [4096] log-probs + [1] loss

  // workspace carve-out (256B aligned)
  char* base = (char*)d_ws;
  size_t off = 0;
  auto alloc = [&](size_t bytes) -> void* {
    void* p = base + off;
    off += (bytes + 255) & ~(size_t)255;
    return p;
  };
  unsigned short* xb      = (unsigned short*)alloc(4096u * 512 * 2);
  unsigned short* W1b     = (unsigned short*)alloc(2048u * 512 * 2);
  unsigned short* W2b     = (unsigned short*)alloc(128u * 2048 * 2);
  unsigned short* W3b     = (unsigned short*)alloc(512u * 128 * 2);
  unsigned short* headWb  = (unsigned short*)alloc(2048u * 512 * 2);   // pad 2002->2048
  unsigned short* t0ab    = (unsigned short*)alloc(128u * 512 * 2);
  unsigned short* t0bb    = (unsigned short*)alloc(8192u * 128 * 2);   // pad 8000->8192
  unsigned short* t1ab    = (unsigned short*)alloc(32u * 512 * 2);
  unsigned short* t1bb    = (unsigned short*)alloc(40448u * 32 * 2);   // pad 40000->40448
  unsigned short* h1b     = (unsigned short*)alloc(4096u * 2048 * 2);
  unsigned short* h2b     = (unsigned short*)alloc(4096u * 128 * 2);
  unsigned short* scoreb  = (unsigned short*)alloc(4096u * 512 * 2);
  unsigned short* u0b     = (unsigned short*)alloc(4096u * 128 * 2);
  unsigned short* u1b     = (unsigned short*)alloc(4096u * 32 * 2);
  float* lse_h   = (float*)alloc(4096u * 4);
  float* lse_t0  = (float*)alloc(4096u * 4);
  float* lse_t1  = (float*)alloc(4096u * 4);
  float* partial = (float*)alloc(512u * 4);

  auto cvt = [&](const float* src, unsigned short* dst, int n, int ntot) {
    cvt_kernel<<<(ntot + 255) / 256, 256, 0, stream>>>(src, dst, n, ntot);
  };
  cvt(x,      xb,     4096 * 512,  4096 * 512);
  cvt(W1,     W1b,    2048 * 512,  2048 * 512);
  cvt(W2,     W2b,    128 * 2048,  128 * 2048);
  cvt(W3,     W3b,    512 * 128,   512 * 128);
  cvt(head_W, headWb, 2002 * 512,  2048 * 512);   // zero-pad rows 2002..2047
  cvt(t0a,    t0ab,   128 * 512,   128 * 512);
  cvt(t0bw,   t0bb,   8000 * 128,  8192 * 128);   // zero-pad rows 8000..8191
  cvt(t1a,    t1ab,   32 * 512,    32 * 512);
  cvt(t1bw,   t1bb,   40000 * 32,  40448 * 32);   // zero-pad rows 40000..40447

  auto gemm = [&](const unsigned short* A, const unsigned short* W,
                  const float* bias, unsigned short* D,
                  int M, int N, int K, int relu) {
    const int strips = (M / 64) * (N / 16);
    gemm_bf16_kernel<<<(strips + 7) / 8, 256, 0, stream>>>(A, W, bias, D, M, N, K, relu);
  };
  gemm(xb,     W1b,  b1,      h1b,    4096, 2048, 512,  1);   // h1 = relu(x@W1^T+b1)
  gemm(h1b,    W2b,  b2,      h2b,    4096, 128,  2048, 1);   // h2
  gemm(h2b,    W3b,  b3,      scoreb, 4096, 512,  128,  1);   // score
  gemm(scoreb, t0ab, nullptr, u0b,    4096, 128,  512,  0);   // u0 = score@t0a^T
  gemm(scoreb, t1ab, nullptr, u1b,    4096, 32,   512,  0);   // u1 = score@t1a^T

  // fused logits + online logsumexp (256 row-blocks each)
  lse_kernel<<<256, 256, 0, stream>>>(scoreb, headWb, head_b, lse_h, 512, 2002, 2048 / 16);
  lse_kernel<<<256, 256, 0, stream>>>(u0b,    t0bb,   nullptr, lse_t0, 128, 8000, 8192 / 16);
  lse_kernel<<<256, 256, 0, stream>>>(u1b,    t1bb,   nullptr, lse_t1, 32, 40000, 40448 / 16);

  final_kernel<<<512, 256, 0, stream>>>(scoreb, u0b, u1b, head_W, head_b,
                                        t0bw, t1bw, t, lse_h, lse_t0, lse_t1,
                                        out, partial);
  loss_kernel<<<1, 256, 0, stream>>>(partial, 512, out + 4096);
}
